// SplineCoupling_7275674600569
// MI455X (gfx1250) — matrix operand order repaired and checked by reading, not compiled, not run on hardware
//
#include <hip/hip_runtime.h>
#include <hip/hip_bf16.h>

// ---------------------------------------------------------------------------
// Fused conditioner-MLP + rational-quadratic spline coupling for gfx1250.
//   B=32768, D1=64, D2=64, C=128, H=128, P=64*49=3136, BINS=16
// GEMMs run on v_wmma_f32_16x16x32_bf16 (fp32 accumulate); the (B,3136) raw
// conditioner output never touches HBM — it is staged per 16-dim chunk in LDS
// and consumed by a single-pass spline (incremental knot capture) immediately.
// Softplus/log use the hardware TRANS pipe (v_exp_f32 / v_log_f32).
// ---------------------------------------------------------------------------

typedef __bf16 v16bf __attribute__((ext_vector_type(16)));
typedef __bf16 v8bf  __attribute__((ext_vector_type(8)));
typedef float  v8f   __attribute__((ext_vector_type(8)));

#define B_TOT   32768
#define D1      64
#define D2      64
#define CDIM    128
#define HDIM    128
#define XDIM    192      // D2 + CDIM
#define PDIM    3136     // D1 * NPAR
#define NBINS   16
#define NPAR    49       // 3*NBINS + 1

// ---- LDS layout (bytes), single 55 KB static block -------------------------
#define A0_OFF   0            // bf16 [16][200]  x-tile (concat(u2,cond))
#define A0_LD    200
#define A1_OFF   6400         // bf16 [16][136]  h1
#define A12_LD   136
#define RAW_OFF  0            // fp32 [16][785]  raw-param stage (reuses A0/A1)
#define RAW_LD   785          // odd stride -> conflict-free rows (17 coprime 64)
#define A2_OFF   50240        // bf16 [16][136]  h2
#define RED_OFF  54592        // fp32 [16][8]    log-jac reduction
#define SMEM_BYTES 55104

// A fragment (16x32 bf16, MxK) for v_wmma_f32_16x16x32_bf16, contiguous K:
//   lanes 0-15 : row m=lane,    e0..e15 = K k0..k0+15
//   lanes 16-31: row m=lane-16, e0..e15 = K k0+16..k0+31
static __device__ __forceinline__ v16bf afrag(const __bf16* base, int ld,
                                              int m, int k0, int lane) {
  const __bf16* p = base + m * ld + k0 + ((lane < 16) ? 0 : 16);
  v8bf lo = *(const v8bf*)(p);
  v8bf hi = *(const v8bf*)(p + 8);
  return __builtin_shufflevector(lo, hi, 0,1,2,3,4,5,6,7,8,9,10,11,12,13,14,15);
}

// B fragment (32x16 bf16, KxN); W stored transposed [N][K] so each lane reads
// 16 contiguous K values of its column n = lane%16 (halves k0 / k0+16).
static __device__ __forceinline__ v16bf bfrag(const __bf16* W, int ldk,
                                              int n, int k0, int lane) {
  const __bf16* p = W + (size_t)n * ldk + k0 + ((lane < 16) ? 0 : 16);
  return *(const v16bf*)p;
}

#define WMMA(a, b, c) \
  __builtin_amdgcn_wmma_f32_16x16x32_bf16(false, (a), false, (b), (short)0, \
                                          (c), false, false)

// softplus on the TRANS pipe: v_exp_f32 + v_log_f32. For x<<0 the absolute
// error vs log1p is <~6e-8 while widths/heights sum to O(1) -> negligible.
static __device__ __forceinline__ float softplusf(float x) {
  return (x > 15.0f) ? x : __logf(1.0f + __expf(x));
}

// ---------------------------------------------------------------------------
// Prep: fp32 weights -> bf16, transposed to [N][K] for the B-fragment layout.
// ---------------------------------------------------------------------------
__global__ __launch_bounds__(256) void prep_weights(
    const float* __restrict__ W1, const float* __restrict__ W2,
    const float* __restrict__ W3,
    __bf16* __restrict__ Wt1, __bf16* __restrict__ Wt2,
    __bf16* __restrict__ Wt3) {
  int i = blockIdx.x * 256 + threadIdx.x;
  if (i < HDIM * XDIM) {                       // Wt1[n][k] = W1[k][n]
    int n = i / XDIM, k = i - n * XDIM;
    Wt1[i] = (__bf16)W1[k * HDIM + n];
  }
  int j = i - HDIM * XDIM;
  if (j >= 0 && j < HDIM * HDIM) {             // Wt2[n][k] = W2[k][n]
    int n = j / HDIM, k = j - n * HDIM;
    Wt2[j] = (__bf16)W2[k * HDIM + n];
  }
  int l = i - HDIM * XDIM - HDIM * HDIM;
  if (l >= 0 && l < PDIM * HDIM) {             // Wt3[n][k] = W3[k][n]
    int n = l / HDIM, k = l - n * HDIM;
    Wt3[l] = (__bf16)W3[(size_t)k * PDIM + n];
  }
}

// ---------------------------------------------------------------------------
// Fused kernel: 128 threads (4 waves), 16 batch rows per workgroup.
// ---------------------------------------------------------------------------
__global__ __launch_bounds__(128) void spline_coupling_fused(
    const float* __restrict__ u1, const float* __restrict__ u2,
    const float* __restrict__ cond,
    const float* __restrict__ b1, const float* __restrict__ b2,
    const float* __restrict__ b3,
    const __bf16* __restrict__ Wt1, const __bf16* __restrict__ Wt2,
    const __bf16* __restrict__ Wt3,
    float* __restrict__ out_v, float* __restrict__ out_lj) {
  __shared__ __align__(16) char smem[SMEM_BYTES];
  __bf16* A0  = (__bf16*)(smem + A0_OFF);
  __bf16* A1  = (__bf16*)(smem + A1_OFF);
  __bf16* A2  = (__bf16*)(smem + A2_OFF);
  float*  RAW = (float*)(smem + RAW_OFF);
  float*  RED = (float*)(smem + RED_OFF);

  const int tid  = threadIdx.x;
  const int lane = tid & 31;
  const int wid  = tid >> 5;
  const int hh   = lane >> 4;   // half-wave select
  const int ln16 = lane & 15;
  const size_t gb0 = (size_t)blockIdx.x * 16;

  // ---- stage x = concat(u2, cond) tile (16 x 192) as bf16 ------------------
  for (int i = tid; i < 16 * XDIM; i += 128) {
    int r = i / XDIM, c = i - r * XDIM;
    size_t gb = gb0 + r;
    float v = (c < D2) ? u2[gb * D2 + c] : cond[gb * CDIM + (c - D2)];
    A0[r * A0_LD + c] = (__bf16)v;
  }
  __syncthreads();

  // ---- GEMM1: h1 = relu(x @ W1 + b1); each wave owns 32 output cols --------
  for (int sub = 0; sub < 2; ++sub) {
    int nb = wid * 32 + sub * 16;
    v8f acc = {};
#pragma unroll
    for (int kc = 0; kc < 6; ++kc) {
      v16bf a = afrag(A0, A0_LD, ln16, kc * 32, lane);
      v16bf b = bfrag(Wt1, XDIM, nb + ln16, kc * 32, lane);
      acc = WMMA(a, b, acc);
    }
    float bias = b1[nb + ln16];
#pragma unroll
    for (int r = 0; r < 8; ++r) {
      int m = hh * 8 + r;   // D layout: lane holds (m = 8*(lane/16)+r, n = lane%16)
      A1[m * A12_LD + nb + ln16] = (__bf16)fmaxf(acc[r] + bias, 0.0f);
    }
  }
  __syncthreads();

  // ---- GEMM2: h2 = relu(h1 @ W2 + b2) --------------------------------------
  for (int sub = 0; sub < 2; ++sub) {
    int nb = wid * 32 + sub * 16;
    v8f acc = {};
#pragma unroll
    for (int kc = 0; kc < 4; ++kc) {
      v16bf a = afrag(A1, A12_LD, ln16, kc * 32, lane);
      v16bf b = bfrag(Wt2, HDIM, nb + ln16, kc * 32, lane);
      acc = WMMA(a, b, acc);
    }
    float bias = b2[nb + ln16];
#pragma unroll
    for (int r = 0; r < 8; ++r) {
      int m = hh * 8 + r;
      A2[m * A12_LD + nb + ln16] = (__bf16)fmaxf(acc[r] + bias, 0.0f);
    }
  }
  __syncthreads();

  // ---- hoist all A fragments of h2 (16x128) once; reused 49x each ----------
  v16bf aF[4];
#pragma unroll
  for (int kc = 0; kc < 4; ++kc) aF[kc] = afrag(A2, A12_LD, ln16, kc * 32, lane);

  const float xshift = logf(expm1f(0.625f));  // default bin width 10/16
  const float dshift = logf(expm1f(1.0f));
  const int   bl  = tid & 15;                 // this thread's fixed batch row
  const int   dlb = tid >> 4;                 // dim-slot base (0..7)
  const size_t gbt = gb0 + bl;
  float ljacc = 0.0f;

  // ---- 4 chunks of 16 dims: GEMM3 tile -> LDS stage -> spline --------------
  // keep the big body un-duplicated: GEMM3 + spline exist once in I$.
#pragma unroll 1
  for (int c = 0; c < 4; ++c) {
    // issue u1 loads early; latency hides under the GEMM3 tile loop
    float tv0 = u1[gbt * D1 + c * 16 + dlb];
    float tv1 = u1[gbt * D1 + c * 16 + dlb + 8];

    __syncthreads();  // previous chunk's spline reads complete

    for (int t = wid; t < 49; t += 4) {
      int n = c * 784 + t * 16 + ln16;
      v8f acc = {};
#pragma unroll
      for (int kc = 0; kc < 4; ++kc) {
        v16bf b = bfrag(Wt3, HDIM, n, kc * 32, lane);
        acc = WMMA(aF[kc], b, acc);
      }
      if (t + 4 < 49)  // warm L2/L0 for this wave's next W3 tile
        __builtin_prefetch(Wt3 + (size_t)(n + 64) * HDIM, 0, 0);
      float bias = b3[n];
      int col = t * 16 + ln16;
#pragma unroll
      for (int r = 0; r < 8; ++r)
        RAW[(hh * 8 + r) * RAW_LD + col] = acc[r] + bias;
    }
    __syncthreads();

    // --- spline over dims [16c, 16c+16): single pass, incremental capture ---
#pragma unroll
    for (int q = 0; q < 2; ++q) {
      const int dl = dlb + q * 8;
      const int d  = c * 16 + dl;
      const float* p  = &RAW[bl * RAW_LD + dl * NPAR];
      const float tv  = (q == 0) ? tv0 : tv1;

      const float left   = p[0] - 5.0f;
      const float bottom = p[1] - 5.0f;

      // cumulative knots; capture knots[lo] / knots[hi] on the fly.
      //   lo = clamp(cnt,1,16)-1 : last knot < tv among knots 0..15
      //   hi = clamp(cnt,1,16)   : first knot >= tv among knots 1..16
      //                            (fallback: knot 16)
      float sw = 0.0f, sh = 0.0f;
      float xk = left, yk = bottom;      // knots[lo]
      float xkp = 0.0f, ykp = 0.0f;      // knots[hi]
      bool  fnd = false;
      int   cnt = (left < tv) ? 1 : 0;   // knots strictly below tv (17 knots)
#pragma unroll
      for (int i = 0; i < NBINS; ++i) {
        sw += softplusf(p[2 + i] + xshift);
        sh += softplusf(p[2 + NBINS + i] + xshift);  // yshift == xshift
        float kx = left + sw;                        // knots_x[i+1]
        float ky = bottom + sh;                      // knots_y[i+1]
        bool below = kx < tv;
        cnt += below ? 1 : 0;
        if (below && i < NBINS - 1) { xk = kx; yk = ky; }   // only knots 1..15
        if (!below && !fnd) { xkp = kx; ykp = ky; fnd = true; }
      }
      if (!fnd) { xkp = left + sw; ykp = bottom + sh; }     // knot 16 fallback

      const bool in_dom = (left < tv) && (tv <= left + sw);
      const int  hi = min(max(cnt, 1), NBINS);
      const int  lo = hi - 1;

      const float scale = sh / sw;
      // der = [scale, softplus(p[34..48]+dshift), scale]; der[j], j in 1..15
      const float dk  = (lo == 0)     ? scale : softplusf(p[33 + lo] + dshift);
      const float dkp = (hi == NBINS) ? scale : softplusf(p[33 + hi] + dshift);

      const float dx  = xkp - xk;
      const float dy  = ykp - yk;
      const float skk = dy / dx;
      float xi = (tv - xk) / dx;
      xi = fminf(fmaxf(xi, 0.0f), 1.0f);
      const float om = 1.0f - xi;

      const float num    = dy * (skk * xi * xi + dk * xi * om);
      const float den    = skk + (dkp + dk - 2.0f * skk) * xi * om;
      const float res_in = yk + num / den;
      const float jnum   = skk * skk *
                           (dkp * xi * xi + 2.0f * skk * xi * om + dk * om * om);
      const float lj_in  = __logf(jnum + 1e-10f) - __logf(den * den + 1e-10f);

      const float res_out = scale * tv + (bottom - scale * left);
      const float lj_out  = __logf(scale + 1e-10f);

      out_v[gbt * D1 + d] = in_dom ? res_in : res_out;
      ljacc += in_dom ? lj_in : lj_out;
    }
  }

  // ---- log_det_J: reduce the 8 per-thread partials of each batch row -------
  __syncthreads();
  RED[bl * 8 + dlb] = ljacc;
  __syncthreads();
  if (tid < 16) {
    float s = 0.0f;
#pragma unroll
    for (int k = 0; k < 8; ++k) s += RED[tid * 8 + k];
    out_lj[gb0 + tid] = s;
  }
}

// ---------------------------------------------------------------------------
extern "C" void kernel_launch(void* const* d_in, const int* in_sizes, int n_in,
                              void* d_out, int out_size, void* d_ws,
                              size_t ws_size, hipStream_t stream) {
  (void)in_sizes; (void)n_in; (void)out_size; (void)ws_size;
  const float* u1   = (const float*)d_in[0];
  const float* u2   = (const float*)d_in[1];
  const float* cond = (const float*)d_in[2];
  const float* W1   = (const float*)d_in[3];
  const float* b1   = (const float*)d_in[4];
  const float* W2   = (const float*)d_in[5];
  const float* b2   = (const float*)d_in[6];
  const float* W3   = (const float*)d_in[7];
  const float* b3   = (const float*)d_in[8];

  float* out_v  = (float*)d_out;                       // (B, 64)
  float* out_lj = out_v + (size_t)B_TOT * D1;          // (B,)

  __bf16* Wt1 = (__bf16*)d_ws;                         // [128][192]
  __bf16* Wt2 = Wt1 + HDIM * XDIM;                     // [128][128]
  __bf16* Wt3 = Wt2 + HDIM * HDIM;                     // [3136][128]

  const int prep_elems = HDIM * XDIM + HDIM * HDIM + PDIM * HDIM;  // 442368
  prep_weights<<<(prep_elems + 255) / 256, 256, 0, stream>>>(W1, W2, W3,
                                                            Wt1, Wt2, Wt3);
  spline_coupling_fused<<<B_TOT / 16, 128, 0, stream>>>(
      u1, u2, cond, b1, b2, b3, Wt1, Wt2, Wt3, out_v, out_lj);
}
